// QuantizedLinear_16836271801128
// MI455X (gfx1250) — matrix operand order repaired and checked by reading, not compiled, not run on hardware
//
#include <hip/hip_runtime.h>

// y = x @ (scale*(q - zp)).T + bias
// f16 WMMA GEMM: y = scale * (x_f16 @ (q - zp)_f16^T) + bias
// (q - zp) is an integer in [-255, 255] -> exact in f16.

typedef __attribute__((ext_vector_type(16))) _Float16 v16h;
typedef __attribute__((ext_vector_type(8)))  _Float16 v8h;
typedef __attribute__((ext_vector_type(4)))  _Float16 v4h;
typedef __attribute__((ext_vector_type(8)))  float    v8f;

#define TILE_M 128
#define TILE_N 128

// ---------------- L2-friendly grouped rasterization ----------------
// Group of GROUP_M block-rows sweeps all block-cols column-major:
// keeps GROUP_M x-strips (16 * 2MB = 32MB) + current weight strip in L2.
__device__ __forceinline__ void tile_coords(int pid, int num_m, int num_n,
                                            int& pm, int& pn) {
    const int GROUP_M = 16;
    const int group_size = GROUP_M * num_n;
    const int group   = pid / group_size;
    const int first_m = group * GROUP_M;
    const int rem     = pid - group * group_size;
    int gm = num_m - first_m; if (gm > GROUP_M) gm = GROUP_M;
    pm = first_m + (rem % gm);
    pn = rem / gm;
}

// =====================================================================
// Path A (fallback): fused dequant GEMM, register-staged global->LDS
// =====================================================================
#define FK_BK 32
#define FK_STRIDE 40  // halves per row: 32 + 8 pad -> 80B (16B aligned)

__global__ __launch_bounds__(256, 2)
void qlinear_wmma_fused(const float* __restrict__ x,
                        const int*   __restrict__ qw,
                        const float* __restrict__ scale_p,
                        const float* __restrict__ zp_p,
                        const float* __restrict__ bias,
                        float*       __restrict__ out,
                        int M, int N, int K)
{
    __shared__ __align__(16) _Float16 sA[2][TILE_M * FK_STRIDE];
    __shared__ __align__(16) _Float16 sB[2][TILE_N * FK_STRIDE];

    const int tid      = threadIdx.x;
    const int lane     = tid & 31;
    const int wave     = tid >> 5;
    const int wm       = wave >> 2;   // 0..1
    const int wn       = wave & 3;    // 0..3
    const int half_sel = lane >> 4;
    const int l16      = lane & 15;

    int pm, pn;
    tile_coords(blockIdx.x, M / TILE_M, N / TILE_N, pm, pn);
    const int m0 = pm * TILE_M;
    const int n0 = pn * TILE_N;

    const float scale = scale_p[0];
    const float zp    = zp_p[0];

    v8f acc[4][2];
#pragma unroll
    for (int mi = 0; mi < 4; ++mi)
#pragma unroll
        for (int ni = 0; ni < 2; ++ni)
#pragma unroll
            for (int v = 0; v < 8; ++v)
                acc[mi][ni][v] = 0.0f;

    float4 ra[4];
    int4   rb[4];
    const int nk = K / FK_BK;

    auto load_tile = [&](int kt) {
#pragma unroll
        for (int i = 0; i < 4; ++i) {
            const int idx = tid + i * 256;
            const int row = idx >> 3;
            const int kv  = idx & 7;
            ra[i] = *(const float4*)(x  + (size_t)(m0 + row) * K + (size_t)kt * FK_BK + kv * 4);
            rb[i] = *(const int4*)  (qw + (size_t)(n0 + row) * K + (size_t)kt * FK_BK + kv * 4);
        }
    };
    auto store_tile = [&](int buf) {
#pragma unroll
        for (int i = 0; i < 4; ++i) {
            const int idx = tid + i * 256;
            const int row = idx >> 3;
            const int kv  = idx & 7;
            v4h ha, hb;
            ha[0] = (_Float16)ra[i].x;  ha[1] = (_Float16)ra[i].y;
            ha[2] = (_Float16)ra[i].z;  ha[3] = (_Float16)ra[i].w;
            hb[0] = (_Float16)((float)rb[i].x - zp);
            hb[1] = (_Float16)((float)rb[i].y - zp);
            hb[2] = (_Float16)((float)rb[i].z - zp);
            hb[3] = (_Float16)((float)rb[i].w - zp);
            *(v4h*)&sA[buf][row * FK_STRIDE + kv * 4] = ha;
            *(v4h*)&sB[buf][row * FK_STRIDE + kv * 4] = hb;
        }
    };
    auto compute = [&](int buf) {
        v16h afrag[4];
#pragma unroll
        for (int mi = 0; mi < 4; ++mi) {
            const _Float16* ap = &sA[buf][(wm * 64 + mi * 16 + l16) * FK_STRIDE + half_sel * 16];
            v8h lo = *(const v8h*)ap;
            v8h hi = *(const v8h*)(ap + 8);
            afrag[mi] = __builtin_shufflevector(lo, hi,
                0,1,2,3,4,5,6,7,8,9,10,11,12,13,14,15);
        }
        v16h bfrag[2];
#pragma unroll
        for (int ni = 0; ni < 2; ++ni) {
            const _Float16* bp = &sB[buf][(wn * 32 + ni * 16 + l16) * FK_STRIDE + half_sel * 16];
            v8h lo = *(const v8h*)bp;
            v8h hi = *(const v8h*)(bp + 8);
            bfrag[ni] = __builtin_shufflevector(lo, hi,
                0,1,2,3,4,5,6,7,8,9,10,11,12,13,14,15);
        }
#pragma unroll
        for (int mi = 0; mi < 4; ++mi)
#pragma unroll
            for (int ni = 0; ni < 2; ++ni)
                acc[mi][ni] = __builtin_amdgcn_wmma_f32_16x16x32_f16(
                    false, afrag[mi], false, bfrag[ni],
                    (short)0, acc[mi][ni], false, false);
    };

    load_tile(0);
    store_tile(0);
    __syncthreads();
    for (int kt = 0; kt < nk; ++kt) {
        const int buf = kt & 1;
        if (kt + 1 < nk) load_tile(kt + 1);
        compute(buf);
        if (kt + 1 < nk) store_tile(buf ^ 1);
        __syncthreads();
    }

#pragma unroll
    for (int mi = 0; mi < 4; ++mi)
#pragma unroll
        for (int ni = 0; ni < 2; ++ni) {
            const int col = n0 + wn * 32 + ni * 16 + l16;
            const float bb = bias[col];
#pragma unroll
            for (int v = 0; v < 8; ++v) {
                const int row = m0 + wm * 64 + mi * 16 + v + 8 * half_sel;
                out[(size_t)row * N + col] = scale * acc[mi][ni][v] + bb;
            }
        }
}

// =====================================================================
// Path B: pre-converted f16 operands + async global->LDS staging
// =====================================================================
__global__ __launch_bounds__(256)
void convert_x_f16(const float* __restrict__ x, _Float16* __restrict__ xf, size_t n)
{
    const size_t i = ((size_t)blockIdx.x * blockDim.x + threadIdx.x) * 4;
    if (i >= n) return;
    float4 v = *(const float4*)(x + i);
    v4h h;
    h[0] = (_Float16)v.x; h[1] = (_Float16)v.y;
    h[2] = (_Float16)v.z; h[3] = (_Float16)v.w;
    *(v4h*)(xf + i) = h;
}

__global__ __launch_bounds__(256)
void convert_w_f16(const int* __restrict__ q, const float* __restrict__ zp_p,
                   _Float16* __restrict__ wf, size_t n)
{
    const size_t i = ((size_t)blockIdx.x * blockDim.x + threadIdx.x) * 4;
    if (i >= n) return;
    const float zp = zp_p[0];
    int4 v = *(const int4*)(q + i);
    v4h h;
    h[0] = (_Float16)((float)v.x - zp);
    h[1] = (_Float16)((float)v.y - zp);
    h[2] = (_Float16)((float)v.z - zp);
    h[3] = (_Float16)((float)v.w - zp);
    *(v4h*)(wf + i) = h;
}

#define AK_BK 64
#define AK_STRIDE 72  // halves per row: 64 + 8 pad -> 144B (16B aligned)

__global__ __launch_bounds__(256, 2)
void qlinear_wmma_async(const _Float16* __restrict__ xf,
                        const _Float16* __restrict__ wf,
                        const float* __restrict__ scale_p,
                        const float* __restrict__ bias,
                        float*       __restrict__ out,
                        int M, int N, int K)
{
    __shared__ __align__(16) _Float16 sA[2][TILE_M * AK_STRIDE];
    __shared__ __align__(16) _Float16 sB[2][TILE_N * AK_STRIDE];

    const int tid      = threadIdx.x;
    const int lane     = tid & 31;
    const int wave     = tid >> 5;
    const int wm       = wave >> 2;   // 0..1
    const int wn       = wave & 3;    // 0..3
    const int half_sel = lane >> 4;
    const int l16      = lane & 15;

    int pm, pn;
    tile_coords(blockIdx.x, M / TILE_M, N / TILE_N, pm, pn);
    const int m0 = pm * TILE_M;
    const int n0 = pn * TILE_N;

    const float scale = scale_p[0];

    v8f acc[4][2];
#pragma unroll
    for (int mi = 0; mi < 4; ++mi)
#pragma unroll
        for (int ni = 0; ni < 2; ++ni)
#pragma unroll
            for (int v = 0; v < 8; ++v)
                acc[mi][ni][v] = 0.0f;

    const int nk = K / AK_BK;
    const unsigned sA_base = (unsigned)(uintptr_t)&sA[0][0];  // low 32 bits = LDS offset
    const unsigned sB_base = (unsigned)(uintptr_t)&sB[0][0];
    const unsigned buf_bytes_A = TILE_M * AK_STRIDE * 2;
    const unsigned buf_bytes_B = TILE_N * AK_STRIDE * 2;

    // Each thread async-copies 4 x 16B chunks of A and of B per K-step.
    auto issue_async = [&](int kt, int buf) {
#pragma unroll
        for (int i = 0; i < 4; ++i) {
            const int idx = tid + i * 256;      // 0..1023
            const int row = idx >> 3;           // 0..127
            const int c   = idx & 7;            // 16B chunk within 64-half row
            const unsigned la = sA_base + buf * buf_bytes_A + row * (AK_STRIDE * 2) + c * 16;
            const unsigned long long ga = (unsigned long long)(uintptr_t)xf
                + ((unsigned long long)(m0 + row) * K + (unsigned long long)kt * AK_BK) * 2
                + (unsigned)(c * 16);
            asm volatile("global_load_async_to_lds_b128 %0, %1, off"
                         :: "v"(la), "v"(ga) : "memory");
            const unsigned lb = sB_base + buf * buf_bytes_B + row * (AK_STRIDE * 2) + c * 16;
            const unsigned long long gb = (unsigned long long)(uintptr_t)wf
                + ((unsigned long long)(n0 + row) * K + (unsigned long long)kt * AK_BK) * 2
                + (unsigned)(c * 16);
            asm volatile("global_load_async_to_lds_b128 %0, %1, off"
                         :: "v"(lb), "v"(gb) : "memory");
        }
    };

    auto compute = [&](int buf) {
#pragma unroll
        for (int kc = 0; kc < 2; ++kc) {        // two K=32 chunks of the 64-wide tile
            v16h afrag[4];
#pragma unroll
            for (int mi = 0; mi < 4; ++mi) {
                const _Float16* ap = &sA[buf][(wm * 64 + mi * 16 + l16) * AK_STRIDE
                                              + kc * 32 + half_sel * 16];
                v8h lo = *(const v8h*)ap;
                v8h hi = *(const v8h*)(ap + 8);
                afrag[mi] = __builtin_shufflevector(lo, hi,
                    0,1,2,3,4,5,6,7,8,9,10,11,12,13,14,15);
            }
            v16h bfrag[2];
#pragma unroll
            for (int ni = 0; ni < 2; ++ni) {
                const _Float16* bp = &sB[buf][(wn * 32 + ni * 16 + l16) * AK_STRIDE
                                              + kc * 32 + half_sel * 16];
                v8h lo = *(const v8h*)bp;
                v8h hi = *(const v8h*)(bp + 8);
                bfrag[ni] = __builtin_shufflevector(lo, hi,
                    0,1,2,3,4,5,6,7,8,9,10,11,12,13,14,15);
            }
#pragma unroll
            for (int mi = 0; mi < 4; ++mi)
#pragma unroll
                for (int ni = 0; ni < 2; ++ni)
                    acc[mi][ni] = __builtin_amdgcn_wmma_f32_16x16x32_f16(
                        false, afrag[mi], false, bfrag[ni],
                        (short)0, acc[mi][ni], false, false);
        }
    };

    issue_async(0, 0);
    asm volatile("s_wait_asynccnt 0x0" ::: "memory");
    __syncthreads();
    for (int kt = 0; kt < nk; ++kt) {
        const int buf = kt & 1;
        if (kt + 1 < nk) issue_async(kt + 1, buf ^ 1);   // overlaps WMMAs below
        compute(buf);
        if (kt + 1 < nk)
            asm volatile("s_wait_asynccnt 0x0" ::: "memory");
        __syncthreads();
    }

#pragma unroll
    for (int mi = 0; mi < 4; ++mi)
#pragma unroll
        for (int ni = 0; ni < 2; ++ni) {
            const int col = n0 + wn * 32 + ni * 16 + l16;
            const float bb = bias[col];
#pragma unroll
            for (int v = 0; v < 8; ++v) {
                const int row = m0 + wm * 64 + mi * 16 + v + 8 * half_sel;
                out[(size_t)row * N + col] = scale * acc[mi][ni][v] + bb;
            }
        }
}

// =====================================================================
extern "C" void kernel_launch(void* const* d_in, const int* in_sizes, int n_in,
                              void* d_out, int out_size, void* d_ws, size_t ws_size,
                              hipStream_t stream) {
    const float* x      = (const float*)d_in[0];
    const int*   qw     = (const int*)  d_in[1];
    const float* scale  = (const float*)d_in[2];
    const float* zp     = (const float*)d_in[3];
    const float* bias   = (const float*)d_in[4];
    float*       out    = (float*)d_out;

    const int N = in_sizes[4];
    const int K = in_sizes[1] / N;
    const int M = in_sizes[0] / K;

    const int num_blocks = (M / TILE_M) * (N / TILE_N);
    const size_t nx = (size_t)M * K;
    const size_t nw = (size_t)N * K;
    const size_t need = (nx + nw) * sizeof(_Float16);

    if (ws_size >= need) {
        _Float16* xf = (_Float16*)d_ws;
        _Float16* wf = xf + nx;
        convert_x_f16<<<(int)(nx / 4 / 256), 256, 0, stream>>>(x, xf, nx);
        convert_w_f16<<<(int)(nw / 4 / 256), 256, 0, stream>>>(qw, zp, wf, nw);
        qlinear_wmma_async<<<num_blocks, 256, 0, stream>>>(xf, wf, scale, bias,
                                                           out, M, N, K);
    } else {
        qlinear_wmma_fused<<<num_blocks, 256, 0, stream>>>(x, qw, scale, zp, bias,
                                                           out, M, N, K);
    }
}
